// MultiHeadSelfAttention_50568944943646
// MI455X (gfx1250) — matrix operand order repaired and checked by reading, not compiled, not run on hardware
//
#include <hip/hip_runtime.h>

// ---------------------------------------------------------------------------
// MHSA for B=2, S=2048, H=16, Dh=64, D=1024 on gfx1250 (CDNA5, wave32).
// All matmuls via v_wmma_f32_16x16x32_bf16 (fp32 accumulate).
// Shared operand tiles double-buffered in LDS via async global->LDS copies.
// Fragment ds_loads batched ahead of WMMA groups to hide LDS latency.
// ---------------------------------------------------------------------------

typedef __bf16 bf16_t;
typedef __attribute__((ext_vector_type(16))) __bf16 v16bf;
typedef __attribute__((ext_vector_type(8)))  __bf16 v8bf;
typedef __attribute__((ext_vector_type(8)))  float  v8f;
typedef __attribute__((ext_vector_type(4)))  int    v4i;

#define DEV __device__ __forceinline__
#define AS1 __attribute__((address_space(1)))
#define AS3 __attribute__((address_space(3)))

#if __has_builtin(__builtin_amdgcn_global_load_async_to_lds_b128)
#define HAVE_ASYNC_LOAD 1
#else
#define HAVE_ASYNC_LOAD 0
#endif
#if __has_builtin(__builtin_amdgcn_s_wait_asynccnt)
#define HAVE_ASYNC_WAIT 1
#else
#define HAVE_ASYNC_WAIT 0
#endif

static constexpr int Sc  = 2048;   // sequence length
static constexpr int Hc  = 16;     // heads
static constexpr int DHc = 64;     // head dim
static constexpr int Dc  = 1024;   // model dim
static constexpr int Tk  = 4096;   // total tokens (B*S)

// Padded LDS row strides (elements).  16B accesses by 16 lanes are
// bank-conflict-free when row stride in banks is an odd multiple of 4:
//   40 elems = 80 B = 20 banks;  72 elems = 144 B = 36 banks.
static constexpr int BSTR = 40;    // for 32-wide (k or key) tiles
static constexpr int KSTR = 72;    // for 64-wide (dh) K tiles

DEV v8bf load8(const bf16_t* p) { return *(const v8bf*)p; }

DEV v16bf cat16(v8bf a, v8bf b) {
  return __builtin_shufflevector(a, b, 0,1,2,3,4,5,6,7,8,9,10,11,12,13,14,15);
}

DEV v8f wmma_bf16(v16bf a, v16bf b, v8f c) {
  return __builtin_amdgcn_wmma_f32_16x16x32_bf16(false, a, false, b,
                                                 (short)0, c, false, false);
}

// A-fragment (16x32): lane<16 row=lane K={0..15}; lane>=16 K={16..31}
DEV v16bf load_afrag(const bf16_t* row_ptr, int hi) {
  const bf16_t* p = row_ptr + hi * 16;
  return cat16(load8(p), load8(p + 8));
}

// 16B global -> LDS stage (async if available)
DEV void stage16(const bf16_t* g, bf16_t* l) {
#if HAVE_ASYNC_LOAD
  __builtin_amdgcn_global_load_async_to_lds_b128((AS1 v4i*)(g), (AS3 v4i*)(l),
                                                 0, 0);
#else
  *(v8bf*)l = load8(g);
#endif
}

DEV void stage_wait() {
#if HAVE_ASYNC_LOAD
#if HAVE_ASYNC_WAIT
  __builtin_amdgcn_s_wait_asynccnt(0);
#else
  asm volatile("s_wait_asynccnt 0x0" ::: "memory");
#endif
#endif
}

// B-fragment read from an LDS tile of 64 rows x 32 k (row stride BSTR)
DEV v16bf bfrag_lds(const bf16_t* tile, int t, int col, int hi) {
  const bf16_t* p = tile + (t * 16 + col) * BSTR + hi * 16;
  return cat16(load8(p), load8(p + 8));
}

// ---------------------------------------------------------------------------
// fp32 -> bf16 conversion (grid-stride)
// ---------------------------------------------------------------------------
__global__ void f32_to_bf16_kernel(const float* __restrict__ in,
                                   bf16_t* __restrict__ out, int n) {
  int i = blockIdx.x * blockDim.x + threadIdx.x;
  int stride = gridDim.x * blockDim.x;
  for (; i < n; i += stride) out[i] = (bf16_t)in[i];
}

// ---------------------------------------------------------------------------
// Stage one 64(n) x 32(k) weight tile: 256 threads x 16B.
// ---------------------------------------------------------------------------
DEV void stage_wtile(const bf16_t* W, int nbase, int k0, bf16_t* dst, int tid) {
  int n   = tid >> 2;           // 0..63
  int off = (tid & 3) * 8;      // 0,8,16,24
  stage16(W + (size_t)(nbase + n) * Dc + k0 + off, dst + n * BSTR + off);
}

// ---------------------------------------------------------------------------
// QKV projection.  Block = 8 waves = 128(M) x 64(N) tile; weight tile shared
// through double-buffered LDS.  q/k -> [B,H,S,Dh] bf16, v -> [B,H,Dh,S].
// ---------------------------------------------------------------------------
__global__ void qkv_proj_kernel(const bf16_t* __restrict__ xb,
                                const bf16_t* __restrict__ Wq, const float* __restrict__ bq,
                                const bf16_t* __restrict__ Wk, const float* __restrict__ bk,
                                const bf16_t* __restrict__ Wv, const float* __restrict__ bv,
                                bf16_t* __restrict__ qb, bf16_t* __restrict__ kb,
                                bf16_t* __restrict__ vtb) {
  __shared__ __align__(16) bf16_t btile[2][64 * BSTR];

  int tid  = threadIdx.x;
  int lane = tid & 31;
  int wave = tid >> 5;
  int col  = lane & 15;
  int hi   = (lane >> 4) & 1;

  int bid  = blockIdx.x;
  int mat  = bid / (32 * 16);            // 0=q 1=k 2=v
  int rem  = bid - mat * (32 * 16);
  int mblk = rem >> 4;                   // 0..31  (128-token block)
  int ngrp = rem & 15;                   // 0..15  (64-feature group)
  int nbase = ngrp * 64;
  int row0  = mblk * 128 + wave * 16;

  const bf16_t* W    = (mat == 0) ? Wq : (mat == 1) ? Wk : Wv;
  const float*  bias = (mat == 0) ? bq : (mat == 1) ? bk : bv;

  const bf16_t* arow = xb + (size_t)(row0 + col) * Dc;

  v8f acc[4] = {v8f{}, v8f{}, v8f{}, v8f{}};
  stage_wtile(W, nbase, 0, btile[0], tid);
  v16bf a_next = load_afrag(arow, hi);

  for (int s = 0; s < 32; ++s) {
    stage_wait();
    __syncthreads();
    if (s + 1 < 32) stage_wtile(W, nbase, (s + 1) * 32, btile[(s + 1) & 1], tid);
    v16bf a = a_next;
    if (s + 1 < 32) a_next = load_afrag(arow + (s + 1) * 32, hi);
    const bf16_t* bt = btile[s & 1];
    // batch all fragment loads ahead of the WMMA group
    v16bf bfr[4];
#pragma unroll
    for (int t = 0; t < 4; ++t) bfr[t] = bfrag_lds(bt, t, col, hi);
#pragma unroll
    for (int t = 0; t < 4; ++t) acc[t] = wmma_bf16(a, bfr[t], acc[t]);
  }

#pragma unroll
  for (int t = 0; t < 4; ++t) {
    int n  = nbase + t * 16 + col;       // output feature
    float bn = bias[n];
    int hh = n >> 6;                     // head
    int d  = n & 63;                     // within-head dim
    if (mat == 2) {
      v8bf pk;
#pragma unroll
      for (int i = 0; i < 8; ++i) pk[i] = (bf16_t)(acc[t][i] + bn);
      int m0 = row0 + hi * 8;
      int b  = m0 >> 11, sp = m0 & 2047;
      *(v8bf*)(vtb + (size_t)((b * Hc + hh) * DHc + d) * Sc + sp) = pk;
    } else {
      bf16_t* dst = (mat == 0) ? qb : kb;
#pragma unroll
      for (int i = 0; i < 8; ++i) {
        int m = row0 + hi * 8 + i;
        int b = m >> 11, sp = m & 2047;
        dst[(size_t)((b * Hc + hh) * Sc + sp) * DHc + d] = (bf16_t)(acc[t][i] + bn);
      }
    }
  }
}

// ---------------------------------------------------------------------------
// Flash attention.  8 waves/block all work on the same (b,h) with different
// 16-query tiles -> K and V^T chunks staged once per block in LDS (double
// buffered, async).  Scores computed as S^T so flash-softmax is lane-local
// except one shfl_xor(16); P^T B-fragment built by a half-wave swap.
// ---------------------------------------------------------------------------
__global__ void attention_kernel(const bf16_t* __restrict__ qb,
                                 const bf16_t* __restrict__ kb,
                                 const bf16_t* __restrict__ vtb,
                                 bf16_t* __restrict__ ctxb) {
  __shared__ __align__(16) bf16_t ktile[2][32 * KSTR];   // 32 keys x 64 dh
  __shared__ __align__(16) bf16_t vtile[2][64 * BSTR];   // 64 d   x 32 keys

  int tid  = threadIdx.x;
  int lane = tid & 31;
  int wave = tid >> 5;
  int h = blockIdx.y, b = blockIdx.z;
  int q0 = (blockIdx.x * 8 + wave) * 16;
  int col = lane & 15;
  int hi  = (lane >> 4) & 1;

  const bf16_t* Q  = qb  + (size_t)(b * Hc + h) * Sc * DHc;
  const bf16_t* K  = kb  + (size_t)(b * Hc + h) * Sc * DHc;
  const bf16_t* Vt = vtb + (size_t)(b * Hc + h) * DHc * Sc;

  // staging coordinates (256 threads x 16B per tile)
  int kr  = tid >> 3, ko = (tid & 7) * 8;   // K tile:  row 0..31, off 0..56
  int vr  = tid >> 2, vo = (tid & 3) * 8;   // Vt tile: row 0..63, off 0..24

  // Q^T B-fragments, reused across all key chunks: dh chunks [0..31], [32..63]
  const bf16_t* qrow = Q + (size_t)(q0 + col) * DHc + hi * 16;
  v16bf bq0 = cat16(load8(qrow),      load8(qrow + 8));
  v16bf bq1 = cat16(load8(qrow + 32), load8(qrow + 40));

  v8f acc[4] = {v8f{}, v8f{}, v8f{}, v8f{}};
  float mrun = -__builtin_inff();
  float lrun = 0.0f;
  const float scale = 0.125f;   // 1/sqrt(64)

  // prologue stage of chunk 0
  stage16(K + (size_t)(0 + kr) * DHc + ko, ktile[0] + kr * KSTR + ko);
  stage16(Vt + (size_t)vr * Sc + 0 + vo,   vtile[0] + vr * BSTR + vo);

  for (int it = 0; it < Sc / 32; ++it) {
    int kc  = it * 32;
    int buf = it & 1;
    stage_wait();
    __syncthreads();
    if (it + 1 < Sc / 32) {
      int nkc = kc + 32, nb = buf ^ 1;
      stage16(K + (size_t)(nkc + kr) * DHc + ko, ktile[nb] + kr * KSTR + ko);
      stage16(Vt + (size_t)vr * Sc + nkc + vo,   vtile[nb] + vr * BSTR + vo);
    }

    // ---- scores^T for two 16-key sub-tiles (operands from LDS, batched) ----
    const bf16_t* kr0 = ktile[buf] + col * KSTR;
    const bf16_t* kr1 = ktile[buf] + (16 + col) * KSTR;
    v16bf ka[4];
    ka[0] = load_afrag(kr0,      hi);
    ka[1] = load_afrag(kr0 + 32, hi);
    ka[2] = load_afrag(kr1,      hi);
    ka[3] = load_afrag(kr1 + 32, hi);
    v8f s0 = {}, s1 = {};
    s0 = wmma_bf16(ka[0], bq0, s0);
    s0 = wmma_bf16(ka[1], bq1, s0);
    s1 = wmma_bf16(ka[2], bq0, s1);
    s1 = wmma_bf16(ka[3], bq1, s1);
#pragma unroll
    for (int i = 0; i < 8; ++i) { s0[i] *= scale; s1[i] *= scale; }

    // ---- online softmax (lanes L and L+16 hold the same query column) ----
    float cm = -__builtin_inff();
#pragma unroll
    for (int i = 0; i < 8; ++i) cm = fmaxf(cm, fmaxf(s0[i], s1[i]));
    cm = fmaxf(cm, __shfl_xor(cm, 16, 32));
    float mnew = fmaxf(mrun, cm);
    float corr = __expf(mrun - mnew);

    v8f p0, p1;
    float ls = 0.0f;
#pragma unroll
    for (int i = 0; i < 8; ++i) {
      p0[i] = __expf(s0[i] - mnew);
      p1[i] = __expf(s1[i] - mnew);
      ls += p0[i] + p1[i];
    }
    ls += __shfl_xor(ls, 16, 32);
    lrun = lrun * corr + ls;
    mrun = mnew;
#pragma unroll
    for (int t = 0; t < 4; ++t)
#pragma unroll
      for (int i = 0; i < 8; ++i) acc[t][i] *= corr;

    // ---- build P^T B-fragment (32 keys x 16 queries) via half-wave swap ----
    float p0p[8], p1p[8];
#pragma unroll
    for (int i = 0; i < 8; ++i) {
      p0p[i] = __shfl_xor((float)p0[i], 16, 32);
      p1p[i] = __shfl_xor((float)p1[i], 16, 32);
    }
    v16bf pf;
#pragma unroll
    for (int i = 0; i < 8; ++i) {
      float f0 = hi ? p1p[i] : (float)p0[i];
      float f1 = hi ? (float)p1[i] : p0p[i];
      pf[i]     = (bf16_t)f0;
      pf[i + 8] = (bf16_t)f1;
    }

    // ---- ctx^T += Vt-chunk * P^T (V^T from LDS, batched loads) ----
    v16bf va[4];
#pragma unroll
    for (int t = 0; t < 4; ++t)
      va[t] = load_afrag(vtile[buf] + (t * 16 + col) * BSTR, hi);
#pragma unroll
    for (int t = 0; t < 4; ++t) acc[t] = wmma_bf16(va[t], pf, acc[t]);
  }

  // ---- normalize + store ctx (back to [B,S,D] layout, contiguous in d) ----
  float inv = 1.0f / lrun;
#pragma unroll
  for (int t = 0; t < 4; ++t) {
    v8bf pk;
#pragma unroll
    for (int i = 0; i < 8; ++i) pk[i] = (bf16_t)(acc[t][i] * inv);
    *(v8bf*)(ctxb + (size_t)(b * Sc + q0 + col) * Dc + h * DHc + t * 16 + hi * 8) = pk;
  }
}

// ---------------------------------------------------------------------------
// Output projection: ctx[4096,1024] @ Wo^T + bo -> fp32 d_out (same blocking)
// ---------------------------------------------------------------------------
__global__ void out_proj_kernel(const bf16_t* __restrict__ ctxb,
                                const bf16_t* __restrict__ Wo,
                                const float* __restrict__ bo,
                                float* __restrict__ out) {
  __shared__ __align__(16) bf16_t btile[2][64 * BSTR];

  int tid  = threadIdx.x;
  int lane = tid & 31;
  int wave = tid >> 5;
  int col  = lane & 15;
  int hi   = (lane >> 4) & 1;

  int mblk = blockIdx.x >> 4;
  int ngrp = blockIdx.x & 15;
  int nbase = ngrp * 64;
  int row0  = mblk * 128 + wave * 16;

  const bf16_t* arow = ctxb + (size_t)(row0 + col) * Dc;

  v8f acc[4] = {v8f{}, v8f{}, v8f{}, v8f{}};
  stage_wtile(Wo, nbase, 0, btile[0], tid);
  v16bf a_next = load_afrag(arow, hi);

  for (int s = 0; s < 32; ++s) {
    stage_wait();
    __syncthreads();
    if (s + 1 < 32) stage_wtile(Wo, nbase, (s + 1) * 32, btile[(s + 1) & 1], tid);
    v16bf a = a_next;
    if (s + 1 < 32) a_next = load_afrag(arow + (s + 1) * 32, hi);
    const bf16_t* bt = btile[s & 1];
    v16bf bfr[4];
#pragma unroll
    for (int t = 0; t < 4; ++t) bfr[t] = bfrag_lds(bt, t, col, hi);
#pragma unroll
    for (int t = 0; t < 4; ++t) acc[t] = wmma_bf16(a, bfr[t], acc[t]);
  }

#pragma unroll
  for (int t = 0; t < 4; ++t) {
    int n = nbase + t * 16 + col;
    float bn = bo[n];
#pragma unroll
    for (int i = 0; i < 8; ++i) {
      int m = row0 + hi * 8 + i;
      out[(size_t)m * Dc + n] = acc[t][i] + bn;
    }
  }
}

// ---------------------------------------------------------------------------
// launch
// ---------------------------------------------------------------------------
extern "C" void kernel_launch(void* const* d_in, const int* in_sizes, int n_in,
                              void* d_out, int out_size, void* d_ws, size_t ws_size,
                              hipStream_t stream) {
  const float* x  = (const float*)d_in[0];
  const float* Wq = (const float*)d_in[1];
  const float* bq = (const float*)d_in[2];
  const float* Wk = (const float*)d_in[3];
  const float* bk = (const float*)d_in[4];
  const float* Wv = (const float*)d_in[5];
  const float* bv = (const float*)d_in[6];
  const float* Wo = (const float*)d_in[7];
  const float* bo = (const float*)d_in[8];

  char* ws = (char*)d_ws;
  const size_t MB = 1u << 20;
  bf16_t* xb   = (bf16_t*)(ws + 0 * MB);    // 8 MB   [4096,1024]
  bf16_t* Wqb  = (bf16_t*)(ws + 8 * MB);    // 2 MB
  bf16_t* Wkb  = (bf16_t*)(ws + 10 * MB);   // 2 MB
  bf16_t* Wvb  = (bf16_t*)(ws + 12 * MB);   // 2 MB
  bf16_t* Wob  = (bf16_t*)(ws + 14 * MB);   // 2 MB
  bf16_t* qbuf = (bf16_t*)(ws + 16 * MB);   // 8 MB   [B,H,S,Dh]
  bf16_t* kbuf = (bf16_t*)(ws + 24 * MB);   // 8 MB   [B,H,S,Dh]
  bf16_t* vtb  = (bf16_t*)(ws + 32 * MB);   // 8 MB   [B,H,Dh,S]
  bf16_t* ctxb = (bf16_t*)(ws + 40 * MB);   // 8 MB   [B,S,D]

  // 1) fp32 -> bf16 conversions
  f32_to_bf16_kernel<<<2048, 256, 0, stream>>>(x,  xb,  Tk * Dc);
  f32_to_bf16_kernel<<<1024, 256, 0, stream>>>(Wq, Wqb, Dc * Dc);
  f32_to_bf16_kernel<<<1024, 256, 0, stream>>>(Wk, Wkb, Dc * Dc);
  f32_to_bf16_kernel<<<1024, 256, 0, stream>>>(Wv, Wvb, Dc * Dc);
  f32_to_bf16_kernel<<<1024, 256, 0, stream>>>(Wo, Wob, Dc * Dc);

  // 2) fused QKV projection: 3 mats x 32 mblocks x 16 ngroups = 1536 blocks
  qkv_proj_kernel<<<1536, 256, 0, stream>>>(xb, Wqb, bq, Wkb, bk, Wvb, bv,
                                            qbuf, kbuf, vtb);

  // 3) flash attention: grid (S/16/8, H, B), 8 query-tile waves per block
  attention_kernel<<<dim3(16, Hc, 2), 256, 0, stream>>>(qbuf, kbuf, vtb, ctxb);

  // 4) output projection: 32 mblocks x 16 ngroups = 512 blocks
  out_proj_kernel<<<512, 256, 0, stream>>>(ctxb, Wob, bo, (float*)d_out);
}
